// MultiHeadAttention_41583873360677
// MI455X (gfx1250) — matrix operand order repaired and checked by reading, not compiled, not run on hardware
//
#include <hip/hip_runtime.h>
#include <hip/hip_bf16.h>

// ---------------------------------------------------------------------------
// GPT-2 multi-head attention on CDNA5 (gfx1250), wave32 + v_wmma_f32_16x16x32_f16
//   B=2, S=2048, nx=768, H=12, d=64
// Pipeline:
//   1) f32 -> f16 convert of hidden; f32 -> f16 *transposed* convert of weights
//   2) QKV GEMM (WMMA, 64x32 per wave), scatter into Q[B,H,S,d], K[B,H,S,d],
//      Vt[B,H,d,S] (f16)
//   3) attention: per (b,h,16-row q block): full 16x2048 score row in LDS (f32),
//      softmax producing a normalized f16 P copy in LDS, P*V via WMMA with
//      ds_load_b128 A-fragments (split-K across wave pairs)
//   4) projection GEMM (WMMA, 64x32 per wave) -> f32 out + bias
// ---------------------------------------------------------------------------

typedef __attribute__((ext_vector_type(16))) _Float16 v16h;
typedef __attribute__((ext_vector_type(8)))  _Float16 v8h;
typedef __attribute__((ext_vector_type(8)))  float    v8f;

union Frag16 { v16h f; v8h h[2]; };

#define WMMA(a, b, c) \
  __builtin_amdgcn_wmma_f32_16x16x32_f16(false, (a), false, (b), (short)0, (c), false, false)

// A-matrix fragment (16x32 f16, row-major source, ld elements).
// Per ISA 7.12.2: lane m = L&15, hi = L>>4; halves 0..7 -> K = kb+8*hi+h,
// halves 8..15 -> K = kb+16+8*hi+(h-8).  Two contiguous 16B loads.
static __device__ inline v16h load_afrag(const _Float16* __restrict__ A, int ld,
                                         int m0, int kb, int lane) {
  const int m  = m0 + (lane & 15);
  const int hi = lane >> 4;
  const _Float16* p = A + (size_t)m * ld + kb + 8 * hi;
  Frag16 u;
  u.h[0] = *(const v8h*)(p);
  u.h[1] = *(const v8h*)(p + 16);
  return u.f;
}

// B-matrix fragment (32x16 f16) from a TRANSPOSED (N-major, [N][K]) source.
// Lane n = L&15, hi = L>>4; halves 0..7 -> K = kb+8*hi+h, 8..15 -> K = kb+16+8*hi+(h-8).
static __device__ inline v16h load_bfrag(const _Float16* __restrict__ Bt, int ld,
                                         int n0, int kb, int lane) {
  const int n  = n0 + (lane & 15);
  const int hi = lane >> 4;
  const _Float16* p = Bt + (size_t)n * ld + kb + 8 * hi;
  Frag16 u;
  u.h[0] = *(const v8h*)(p);
  u.h[1] = *(const v8h*)(p + 16);
  return u.f;
}

// A-fragment straight from an f16 LDS tile (row-major, ld elements).
static __device__ inline v16h load_afrag_lds(const _Float16* p16, int ld,
                                             int kb, int lane) {
  const int m  = lane & 15;
  const int hi = lane >> 4;
  const _Float16* p = p16 + (size_t)m * ld + kb + 8 * hi;
  Frag16 u;
  u.h[0] = *(const v8h*)(p);
  u.h[1] = *(const v8h*)(p + 16);
  return u.f;
}

// ---------------------------------------------------------------------------
// Conversion kernels
// ---------------------------------------------------------------------------
__global__ void k_cvt_f16(const float* __restrict__ in, _Float16* __restrict__ out, int n) {
  int i = blockIdx.x * blockDim.x + threadIdx.x;
  if (i < n) out[i] = (_Float16)in[i];
}

// in: [R][C] f32 row-major  ->  out: [C][R] f16 row-major (transpose)
__global__ void k_cvt_T_f16(const float* __restrict__ in, _Float16* __restrict__ out,
                            int R, int C) {
  int i = blockIdx.x * blockDim.x + threadIdx.x;
  if (i < R * C) {
    int r = i / C, c = i % C;
    out[(size_t)c * R + r] = (_Float16)in[i];
  }
}

// ---------------------------------------------------------------------------
// QKV GEMM: [4096 x 768] @ [768 x 2304]  (B operand pre-transposed)
// grid (64, 9), block 256 (8 waves); each wave: 64(M) x 32(N): 8 WMMA / 6 loads
// per K-step.  Epilogue scatters into Q/K/Vt head layouts with bias.
// ---------------------------------------------------------------------------
__global__ void __launch_bounds__(256) k_gemm_qkv(
    const _Float16* __restrict__ Xh, const _Float16* __restrict__ WaT,
    const float* __restrict__ b_attn,
    _Float16* __restrict__ Q, _Float16* __restrict__ K, _Float16* __restrict__ Vt) {
  const int lane = threadIdx.x & 31;
  const int wave = threadIdx.x >> 5;
  const int m0 = blockIdx.x * 64;
  const int n0 = blockIdx.y * 256 + wave * 32;

  v8f c[4][2] = {};
  for (int kb = 0; kb < 768; kb += 32) {
    v16h a0 = load_afrag(Xh, 768, m0, kb, lane);
    v16h a1 = load_afrag(Xh, 768, m0 + 16, kb, lane);
    v16h a2 = load_afrag(Xh, 768, m0 + 32, kb, lane);
    v16h a3 = load_afrag(Xh, 768, m0 + 48, kb, lane);
    v16h b0 = load_bfrag(WaT, 768, n0, kb, lane);
    v16h b1 = load_bfrag(WaT, 768, n0 + 16, kb, lane);
    c[0][0] = WMMA(a0, b0, c[0][0]);
    c[0][1] = WMMA(a0, b1, c[0][1]);
    c[1][0] = WMMA(a1, b0, c[1][0]);
    c[1][1] = WMMA(a1, b1, c[1][1]);
    c[2][0] = WMMA(a2, b0, c[2][0]);
    c[2][1] = WMMA(a2, b1, c[2][1]);
    c[3][0] = WMMA(a3, b0, c[3][0]);
    c[3][1] = WMMA(a3, b1, c[3][1]);
  }

  const int hi = lane >> 4;
  const int ln = lane & 15;
#pragma unroll
  for (int mi = 0; mi < 4; ++mi) {
#pragma unroll
    for (int ni = 0; ni < 2; ++ni) {
      const int gn = n0 + ni * 16 + ln;
      const float bias = b_attn[gn];
#pragma unroll
      for (int r = 0; r < 8; ++r) {
        const int gm = m0 + mi * 16 + r + 8 * hi;   // row in [0,4096)
        const int bb = gm >> 11;                    // batch
        const int s  = gm & 2047;                   // sequence position
        const _Float16 hv = (_Float16)(c[mi][ni][r] + bias);
        if (gn < 768) {
          const int h = gn >> 6, d = gn & 63;
          Q[(((size_t)(bb * 12 + h)) * 2048 + s) * 64 + d] = hv;
        } else if (gn < 1536) {
          const int q2 = gn - 768;
          const int h = q2 >> 6, d = q2 & 63;
          K[(((size_t)(bb * 12 + h)) * 2048 + s) * 64 + d] = hv;
        } else {
          const int q2 = gn - 1536;
          const int h = q2 >> 6, d = q2 & 63;
          Vt[(((size_t)(bb * 12 + h)) * 64 + d) * 2048 + s] = hv;  // transposed V
        }
      }
    }
  }
}

// ---------------------------------------------------------------------------
// Attention: one block per (b, h, 16-query-row tile). 8 waves.
//   Phase 1: scores 16x2048 -> LDS f32, WMMA QK^T, scale 1/sqrt(64)
//   Phase 2: softmax; normalized probabilities written ONCE as f16 into LDS
//   Phase 3: P @ V with ds_load_b128 A-fragments (split-K across wave halves),
//            reduce in LDS, store f16 AO
// LDS: 128KB scores + 64KB P16 + 1KB reduce = 193KB (CDNA5: 320KB/WG)
// ---------------------------------------------------------------------------
__global__ void __launch_bounds__(256) k_attn(
    const _Float16* __restrict__ Q, const _Float16* __restrict__ K,
    const _Float16* __restrict__ Vt, _Float16* __restrict__ AO) {
  __shared__ float    sc[16 * 2048];    // 128 KB raw scores / scratch
  __shared__ _Float16 p16[16 * 2048];   //  64 KB normalized probabilities
  __shared__ float    red[256];

  const int lane = threadIdx.x & 31;
  const int wave = threadIdx.x >> 5;
  const int hi = lane >> 4;
  const int ln = lane & 15;

  const int bid  = blockIdx.x;
  const int qblk = bid & 127;
  const int hh   = (bid >> 7) % 12;
  const int bb   = bid / (128 * 12);

  const _Float16* Qb = Q + (((size_t)(bb * 12 + hh)) * 2048 + qblk * 16) * 64;
  const _Float16* Kb = K + ((size_t)(bb * 12 + hh)) * 2048 * 64;
  const _Float16* Vb = Vt + ((size_t)(bb * 12 + hh)) * 64 * 2048;

  // ---- Phase 1: scores -------------------------------------------------
  v16h qa0 = load_afrag(Qb, 64, 0, 0, lane);
  v16h qa1 = load_afrag(Qb, 64, 0, 32, lane);
  for (int kt = wave; kt < 128; kt += 8) {
    v16h kb0 = load_bfrag(Kb, 64, kt * 16, 0, lane);
    v16h kb1 = load_bfrag(Kb, 64, kt * 16, 32, lane);
    v8f c = {};
    c = WMMA(qa0, kb0, c);
    c = WMMA(qa1, kb1, c);
#pragma unroll
    for (int r = 0; r < 8; ++r)
      sc[(r + 8 * hi) * 2048 + kt * 16 + ln] = c[r] * 0.125f;  // 1/sqrt(64)
  }
  __syncthreads();

  // ---- Phase 2: softmax (writes normalized f16 P exactly once) ---------
  const int row = threadIdx.x >> 4;
  const int sub = threadIdx.x & 15;
  float*    srow = sc  + row * 2048 + sub * 128;
  _Float16* prow = p16 + row * 2048 + sub * 128;

  float m = -3.4e38f;
  for (int i = 0; i < 128; ++i) m = fmaxf(m, srow[i]);
  red[threadIdx.x] = m;
  __syncthreads();
  float rm = -3.4e38f;
  for (int j = 0; j < 16; ++j) rm = fmaxf(rm, red[row * 16 + j]);
  __syncthreads();

  float ssum = 0.f;
  for (int i = 0; i < 128; ++i) {
    float e = __expf(srow[i] - rm);
    srow[i] = e;
    ssum += e;
  }
  red[threadIdx.x] = ssum;
  __syncthreads();
  float rs = 0.f;
  for (int j = 0; j < 16; ++j) rs += red[row * 16 + j];
  const float inv = 1.f / rs;
  for (int i = 0; i < 128; ++i) prow[i] = (_Float16)(srow[i] * inv);
  __syncthreads();

  // ---- Phase 3: P @ V --------------------------------------------------
  const int dtile = wave & 3;       // which 16-wide slice of d=64
  const int half  = wave >> 2;      // split-K half
  v8f acc = {};
  for (int kb = half * 1024; kb < half * 1024 + 1024; kb += 32) {
    v16h pa = load_afrag_lds(p16, 2048, kb, lane);      // two ds_load_b128
    v16h vb = load_bfrag(Vb, 2048, dtile * 16, kb, lane);
    acc = WMMA(pa, vb, acc);
  }
  __syncthreads();                  // everyone done reading p16 -> reuse sc
  float* part = sc;
  if (half == 1) {
#pragma unroll
    for (int r = 0; r < 8; ++r) part[((size_t)dtile * 32 + lane) * 8 + r] = acc[r];
  }
  __syncthreads();
  if (half == 0) {
#pragma unroll
    for (int r = 0; r < 8; ++r) acc[r] += part[((size_t)dtile * 32 + lane) * 8 + r];
#pragma unroll
    for (int r = 0; r < 8; ++r) {
      const int M = r + 8 * hi;
      const int s = qblk * 16 + M;
      AO[((size_t)(bb * 2048 + s)) * 768 + hh * 64 + dtile * 16 + ln] = (_Float16)acc[r];
    }
  }
}

// ---------------------------------------------------------------------------
// Output projection: [4096 x 768] @ [768 x 768] + bias -> f32 out
// grid (64, 3), block 256; wave tile 64x32.
// ---------------------------------------------------------------------------
__global__ void __launch_bounds__(256) k_gemm_proj(
    const _Float16* __restrict__ AO, const _Float16* __restrict__ WpT,
    const float* __restrict__ b_proj, float* __restrict__ out) {
  const int lane = threadIdx.x & 31;
  const int wave = threadIdx.x >> 5;
  const int m0 = blockIdx.x * 64;
  const int n0 = blockIdx.y * 256 + wave * 32;

  v8f c[4][2] = {};
  for (int kb = 0; kb < 768; kb += 32) {
    v16h a0 = load_afrag(AO, 768, m0, kb, lane);
    v16h a1 = load_afrag(AO, 768, m0 + 16, kb, lane);
    v16h a2 = load_afrag(AO, 768, m0 + 32, kb, lane);
    v16h a3 = load_afrag(AO, 768, m0 + 48, kb, lane);
    v16h b0 = load_bfrag(WpT, 768, n0, kb, lane);
    v16h b1 = load_bfrag(WpT, 768, n0 + 16, kb, lane);
    c[0][0] = WMMA(a0, b0, c[0][0]);
    c[0][1] = WMMA(a0, b1, c[0][1]);
    c[1][0] = WMMA(a1, b0, c[1][0]);
    c[1][1] = WMMA(a1, b1, c[1][1]);
    c[2][0] = WMMA(a2, b0, c[2][0]);
    c[2][1] = WMMA(a2, b1, c[2][1]);
    c[3][0] = WMMA(a3, b0, c[3][0]);
    c[3][1] = WMMA(a3, b1, c[3][1]);
  }

  const int hi = lane >> 4;
  const int ln = lane & 15;
#pragma unroll
  for (int mi = 0; mi < 4; ++mi) {
#pragma unroll
    for (int ni = 0; ni < 2; ++ni) {
      const int gn = n0 + ni * 16 + ln;
      const float bias = b_proj[gn];
#pragma unroll
      for (int r = 0; r < 8; ++r) {
        const int gm = m0 + mi * 16 + r + 8 * hi;
        out[(size_t)gm * 768 + gn] = c[mi][ni][r] + bias;
      }
    }
  }
}

// ---------------------------------------------------------------------------
// Host launch
// ---------------------------------------------------------------------------
extern "C" void kernel_launch(void* const* d_in, const int* in_sizes, int n_in,
                              void* d_out, int out_size, void* d_ws, size_t ws_size,
                              hipStream_t stream) {
  (void)in_sizes; (void)n_in; (void)out_size; (void)ws_size;
  const float* hs      = (const float*)d_in[0];   // [2,2048,768]
  const float* w_attn  = (const float*)d_in[1];   // [768,2304]
  const float* b_attn  = (const float*)d_in[2];   // [2304]
  const float* w_proj  = (const float*)d_in[3];   // [768,768]
  const float* b_proj  = (const float*)d_in[4];   // [768]
  float* out = (float*)d_out;                     // [2,2048,768]

  char* ws = (char*)d_ws;
  size_t off = 0;
  auto alloc = [&](size_t bytes) -> void* {
    void* p = ws + off;
    off = (off + bytes + 255) & ~(size_t)255;
    return p;
  };

  const size_t BS = 4096;                 // B*S
  _Float16* Xh  = (_Float16*)alloc(BS * 768 * 2);
  _Float16* WaT = (_Float16*)alloc((size_t)2304 * 768 * 2);
  _Float16* WpT = (_Float16*)alloc((size_t)768 * 768 * 2);
  _Float16* Qb  = (_Float16*)alloc((size_t)2 * 12 * 2048 * 64 * 2);
  _Float16* Kb  = (_Float16*)alloc((size_t)2 * 12 * 2048 * 64 * 2);
  _Float16* Vt  = (_Float16*)alloc((size_t)2 * 12 * 64 * 2048 * 2);
  _Float16* AO  = (_Float16*)alloc(BS * 768 * 2);

  // 1) precision conversion (+ weight transposes)
  k_cvt_f16<<<(BS * 768) / 256, 256, 0, stream>>>(hs, Xh, (int)(BS * 768));
  k_cvt_T_f16<<<(768 * 2304) / 256, 256, 0, stream>>>(w_attn, WaT, 768, 2304);
  k_cvt_T_f16<<<(768 * 768) / 256, 256, 0, stream>>>(w_proj, WpT, 768, 768);

  // 2) QKV GEMM -> Q, K, Vt
  k_gemm_qkv<<<dim3(64, 9), 256, 0, stream>>>(Xh, WaT, b_attn, Qb, Kb, Vt);

  // 3) attention -> AO (f16 [4096, 768])
  k_attn<<<2 * 12 * 128, 256, 0, stream>>>(Qb, Kb, Vt, AO);

  // 4) projection -> out (f32) + bias
  k_gemm_proj<<<dim3(64, 3), 256, 0, stream>>>(AO, WpT, b_proj, out);
}
